// GraphSAGE_12232066859619
// MI455X (gfx1250) — matrix operand order repaired and checked by reading, not compiled, not run on hardware
//
#include <hip/hip_runtime.h>

typedef __attribute__((ext_vector_type(16))) _Float16 v16h;
typedef __attribute__((ext_vector_type(8)))  _Float16 v8h;
typedef __attribute__((ext_vector_type(8)))  float    v8f;

#define N_NODES 50000
#define F_IN    128
#define HID     256
#define CLS     47
#define CLSP    48
#define N_EDGES 625000

// ---------------- utility kernels ----------------

__global__ void k_zero(float* __restrict__ p, int n) {
  int i = blockIdx.x * blockDim.x + threadIdx.x;
  if (i < n) p[i] = 0.0f;
}

__global__ void k_f32_to_f16(const float* __restrict__ in, _Float16* __restrict__ out, int n) {
  int i = blockIdx.x * blockDim.x + threadIdx.x;
  if (i < n) out[i] = (_Float16)in[i];
}

// W [K][Nn] f32 row-major -> Wt [Npad][K] f16 (transposed, pad rows zeroed)
__global__ void k_wt(const float* __restrict__ W, _Float16* __restrict__ Wt,
                     int K, int Nn, int Npad) {
  int i = blockIdx.x * blockDim.x + threadIdx.x;
  int total = Npad * K;
  if (i >= total) return;
  int n = i / K, k = i - n * K;
  Wt[i] = (n < Nn) ? (_Float16)W[k * Nn + n] : (_Float16)0.0f;
}

// mean = (f32)agg_f16 / max(deg,1), converted back to f16
__global__ void k_mean(const _Float16* __restrict__ agg, const float* __restrict__ deg,
                       _Float16* __restrict__ out, int feat, int total) {
  int i = blockIdx.x * blockDim.x + threadIdx.x;
  if (i >= total) return;
  int row = i / feat;
  float d = deg[row];
  d = d > 1.0f ? d : 1.0f;
  out[i] = (_Float16)((float)agg[i] / d);
}

// ---------------- packed f16 atomic (CDNA5 GLOBAL_ATOMIC_PK_ADD_F16) ----------------

__device__ __forceinline__ void pk_atomic_add_f16(_Float16* p, unsigned int pk2) {
  asm volatile("global_atomic_pk_add_f16 %0, %1, off"
               :: "v"((unsigned long long)(uintptr_t)p), "v"(pk2)
               : "memory");
}

// ---------------- edge scatter kernels ----------------

// One wave per edge; lane l handles features [4l,4l+3] of the 128-wide f16 x.
// 2 packed-f16 atomics per lane (64 per edge) instead of 128 f32 atomics.
__global__ void k_scatter1(const _Float16* __restrict__ xh, const int* __restrict__ ei,
                           _Float16* __restrict__ agg, float* __restrict__ deg) {
  int t = blockIdx.x * blockDim.x + threadIdx.x;
  int e = t >> 5;
  int lane = t & 31;
  if (e >= N_EDGES) return;
  int src = ei[e];
  int dst = ei[N_EDGES + e];
  uint2 v = ((const uint2*)(xh + (size_t)src * F_IN))[lane];   // 4 halves
  _Float16* ap = agg + (size_t)dst * F_IN + lane * 4;
  pk_atomic_add_f16(ap + 0, v.x);
  pk_atomic_add_f16(ap + 2, v.y);
  if (lane == 0) atomicAdd(deg + dst, 1.0f);
}

// One wave per edge; lane l handles 8 f16 features of the 256-wide h.
__global__ void k_scatter2(const _Float16* __restrict__ h, const int* __restrict__ ei,
                           _Float16* __restrict__ agg) {
  int t = blockIdx.x * blockDim.x + threadIdx.x;
  int e = t >> 5;
  int lane = t & 31;
  if (e >= N_EDGES) return;
  int src = ei[e];
  int dst = ei[N_EDGES + e];
  uint4 m = ((const uint4*)(h + (size_t)src * HID))[lane];     // 8 halves
  _Float16* ap = agg + (size_t)dst * HID + lane * 8;
  pk_atomic_add_f16(ap + 0, m.x);
  pk_atomic_add_f16(ap + 2, m.y);
  pk_atomic_add_f16(ap + 4, m.z);
  pk_atomic_add_f16(ap + 6, m.w);
}

// ---------------- WMMA fragment loaders ----------------

// A 16x32 f16 fragment: lane holds K chunks [base..base+7] and [base+16..base+23]
__device__ __forceinline__ v16h load_a_frag(const _Float16* base) {
  union { v16h v; v8h h[2]; } u;
  u.h[0] = *(const v8h*)(base);
  u.h[1] = *(const v8h*)(base + 16);
  return u.v;
}

// B 32x16 f16 fragment (from [N][K] pre-transposed weights): lane holds K chunks [base..base+7] and [base+16..base+23]
__device__ __forceinline__ v16h load_b_frag(const _Float16* base) {
  union { v16h v; v8h h[2]; } u;
  u.h[0] = *(const v8h*)(base);
  u.h[1] = *(const v8h*)(base + 16);
  return u.v;
}

__device__ __forceinline__ v8f wmma_f16(v16h a, v16h b, v8f c) {
  return __builtin_amdgcn_wmma_f32_16x16x32_f16(false, a, false, b, (short)0, c, false, false);
}

// ---------------- fused SAGE GEMM layer 1 ----------------
// h = relu(mean1 @ W1l + b1 + x @ W1r). Block = 128 thr = 4 waves; each wave
// computes a 16x64 output strip (4 accumulators reuse each A fragment 4x).
__global__ void k_gemm1(const _Float16* __restrict__ meanA, const _Float16* __restrict__ xA,
                        const _Float16* __restrict__ Wl, const _Float16* __restrict__ Wr,
                        const float* __restrict__ bias, _Float16* __restrict__ h_out) {
  int wave = threadIdx.x >> 5;
  int lane = threadIdx.x & 31;
  int tileM = blockIdx.x * 16;
  int baseN = wave * 64;
  int r16  = lane & 15;
  int half = lane >> 4;

  const _Float16* aMean = meanA + (size_t)(tileM + r16) * F_IN + half * 8;
  const _Float16* aX    = xA    + (size_t)(tileM + r16) * F_IN + half * 8;
  const _Float16* bl    = Wl + (size_t)(baseN + r16) * F_IN + half * 8;
  const _Float16* br    = Wr + (size_t)(baseN + r16) * F_IN + half * 8;

  v8f acc[4] = {};
#pragma unroll
  for (int k0 = 0; k0 < F_IN; k0 += 32) {
    v16h am = load_a_frag(aMean + k0);
#pragma unroll
    for (int nt = 0; nt < 4; ++nt)
      acc[nt] = wmma_f16(am, load_b_frag(bl + (size_t)nt * 16 * F_IN + k0), acc[nt]);
    v16h ax = load_a_frag(aX + k0);
#pragma unroll
    for (int nt = 0; nt < 4; ++nt)
      acc[nt] = wmma_f16(ax, load_b_frag(br + (size_t)nt * 16 * F_IN + k0), acc[nt]);
  }

  _Float16* obRow = h_out + (size_t)(tileM + half * 8) * HID;
#pragma unroll
  for (int nt = 0; nt < 4; ++nt) {
    int col = baseN + nt * 16 + r16;
    float bv = bias[col];
#pragma unroll
    for (int r = 0; r < 8; ++r) {
      float v = acc[nt][r] + bv;
      v = v > 0.0f ? v : 0.0f;
      obRow[(size_t)r * HID + col] = (_Float16)v;
    }
  }
}

// ---------------- fused SAGE GEMM layer 2 ----------------
// logits[:,48] = mean2 @ W2l + b2 + h @ W2r (col 47 is zero-weight pad).
// Block = 160 thr = 5 waves; each wave does one 16-row tile x all 48 cols.
// 3125 M-tiles = 625 blocks * 5 waves exactly.
__global__ void k_gemm2(const _Float16* __restrict__ meanA, const _Float16* __restrict__ hA,
                        const _Float16* __restrict__ Wl, const _Float16* __restrict__ Wr,
                        const float* __restrict__ bias, float* __restrict__ logits) {
  int wave = threadIdx.x >> 5;
  int lane = threadIdx.x & 31;
  int tileM = (blockIdx.x * 5 + wave) * 16;
  int r16  = lane & 15;
  int half = lane >> 4;

  const _Float16* aMean = meanA + (size_t)(tileM + r16) * HID + half * 8;
  const _Float16* aH    = hA    + (size_t)(tileM + r16) * HID + half * 8;
  const _Float16* bl    = Wl + (size_t)r16 * HID + half * 8;
  const _Float16* br    = Wr + (size_t)r16 * HID + half * 8;

  v8f acc[3] = {};
#pragma unroll
  for (int k0 = 0; k0 < HID; k0 += 32) {
    v16h am = load_a_frag(aMean + k0);
#pragma unroll
    for (int nt = 0; nt < 3; ++nt)
      acc[nt] = wmma_f16(am, load_b_frag(bl + (size_t)nt * 16 * HID + k0), acc[nt]);
    v16h ah = load_a_frag(aH + k0);
#pragma unroll
    for (int nt = 0; nt < 3; ++nt)
      acc[nt] = wmma_f16(ah, load_b_frag(br + (size_t)nt * 16 * HID + k0), acc[nt]);
  }

  float* obRow = logits + (size_t)(tileM + half * 8) * CLSP;
#pragma unroll
  for (int nt = 0; nt < 3; ++nt) {
    int col = nt * 16 + r16;
    float bv = (col < CLS) ? bias[col] : 0.0f;
#pragma unroll
    for (int r = 0; r < 8; ++r)
      obRow[(size_t)r * CLSP + col] = acc[nt][r] + bv;
  }
}

// ---------------- log_softmax over 47 classes, one wave per row ----------------
__global__ void k_logsoftmax(const float* __restrict__ logits, float* __restrict__ out) {
  int row = blockIdx.x * blockDim.y + threadIdx.y;
  int lane = threadIdx.x;
  if (row >= N_NODES) return;
  const float* lp = logits + (size_t)row * CLSP;
  float v0 = (lane < CLS)      ? lp[lane]      : -1e30f;
  float v1 = (lane + 32 < CLS) ? lp[lane + 32] : -1e30f;
  float m = fmaxf(v0, v1);
#pragma unroll
  for (int off = 16; off > 0; off >>= 1) m = fmaxf(m, __shfl_xor(m, off, 32));
  float s = 0.0f;
  if (lane < CLS)      s += expf(v0 - m);
  if (lane + 32 < CLS) s += expf(v1 - m);
#pragma unroll
  for (int off = 16; off > 0; off >>= 1) s += __shfl_xor(s, off, 32);
  float lse = m + logf(s);
  if (lane < CLS)      out[(size_t)row * CLS + lane]      = v0 - lse;
  if (lane + 32 < CLS) out[(size_t)row * CLS + lane + 32] = v1 - lse;
}

// ---------------- launch ----------------

static inline size_t align256(size_t x) { return (x + 255) & ~(size_t)255; }

extern "C" void kernel_launch(void* const* d_in, const int* in_sizes, int n_in,
                              void* d_out, int out_size, void* d_ws, size_t ws_size,
                              hipStream_t stream) {
  (void)in_sizes; (void)n_in; (void)out_size; (void)ws_size;

  const float* x    = (const float*)d_in[0];
  const int*   ei   = (const int*)  d_in[1];
  const float* W1l  = (const float*)d_in[2];
  const float* b1   = (const float*)d_in[3];
  const float* W1r  = (const float*)d_in[4];
  const float* W2l  = (const float*)d_in[5];
  const float* b2   = (const float*)d_in[6];
  const float* W2r  = (const float*)d_in[7];
  float* out = (float*)d_out;

  // carve workspace
  char* w = (char*)d_ws;
  size_t off = 0;
  float*    deg     = (float*)(w + off);    off = align256(off + (size_t)N_NODES * 4);
  _Float16* agg1    = (_Float16*)(w + off); off = align256(off + (size_t)N_NODES * F_IN * 2);
  _Float16* agg2    = (_Float16*)(w + off); off = align256(off + (size_t)N_NODES * HID * 2);
  _Float16* x_h     = (_Float16*)(w + off); off = align256(off + (size_t)N_NODES * F_IN * 2);
  _Float16* mean1_h = (_Float16*)(w + off); off = align256(off + (size_t)N_NODES * F_IN * 2);
  _Float16* h_h     = (_Float16*)(w + off); off = align256(off + (size_t)N_NODES * HID * 2);
  _Float16* mean2_h = (_Float16*)(w + off); off = align256(off + (size_t)N_NODES * HID * 2);
  float*    logits  = (float*)(w + off);    off = align256(off + (size_t)N_NODES * CLSP * 4);
  _Float16* W1l_t   = (_Float16*)(w + off); off = align256(off + (size_t)HID * F_IN * 2);
  _Float16* W1r_t   = (_Float16*)(w + off); off = align256(off + (size_t)HID * F_IN * 2);
  _Float16* W2l_t   = (_Float16*)(w + off); off = align256(off + (size_t)CLSP * HID * 2);
  _Float16* W2r_t   = (_Float16*)(w + off); off = align256(off + (size_t)CLSP * HID * 2);

  const int B = 256;
  // zero accumulators (workspace is poisoned; must re-zero every call).
  // f16 aggregate buffers zeroed as dwords (0x00000000 == two +0.0h).
  {
    int n = N_NODES;
    k_zero<<<(n + B - 1) / B, B, 0, stream>>>(deg, n);
    n = N_NODES * F_IN / 2;
    k_zero<<<(n + B - 1) / B, B, 0, stream>>>((float*)agg1, n);
    n = N_NODES * HID / 2;
    k_zero<<<(n + B - 1) / B, B, 0, stream>>>((float*)agg2, n);
  }

  // precision prep: f16 copy of x, transposed f16 weights
  {
    int n = N_NODES * F_IN;
    k_f32_to_f16<<<(n + B - 1) / B, B, 0, stream>>>(x, x_h, n);
    int t1 = HID * F_IN;
    k_wt<<<(t1 + B - 1) / B, B, 0, stream>>>(W1l, W1l_t, F_IN, HID, HID);
    k_wt<<<(t1 + B - 1) / B, B, 0, stream>>>(W1r, W1r_t, F_IN, HID, HID);
    int t2 = CLSP * HID;
    k_wt<<<(t2 + B - 1) / B, B, 0, stream>>>(W2l, W2l_t, HID, CLS, CLSP);
    k_wt<<<(t2 + B - 1) / B, B, 0, stream>>>(W2r, W2r_t, HID, CLS, CLSP);
  }

  // layer 1: aggregate (packed f16 atomics), mean, fused WMMA GEMM + bias + relu
  {
    int threads = N_EDGES * 32;
    k_scatter1<<<(threads + B - 1) / B, B, 0, stream>>>(x_h, ei, agg1, deg);
    int n = N_NODES * F_IN;
    k_mean<<<(n + B - 1) / B, B, 0, stream>>>(agg1, deg, mean1_h, F_IN, n);
    k_gemm1<<<N_NODES / 16, 128, 0, stream>>>(mean1_h, x_h, W1l_t, W1r_t, b1, h_h);
  }

  // layer 2: aggregate, mean, fused WMMA GEMM + bias
  {
    int threads = N_EDGES * 32;
    k_scatter2<<<(threads + B - 1) / B, B, 0, stream>>>(h_h, ei, agg2);
    int n = N_NODES * HID;
    k_mean<<<(n + B - 1) / B, B, 0, stream>>>(agg2, deg, mean2_h, HID, n);
    k_gemm2<<<N_NODES / 16 / 5, 160, 0, stream>>>(mean2_h, h_h, W2l_t, W2r_t, b2, logits);
  }

  // log_softmax -> output
  {
    dim3 blk(32, 8);
    k_logsoftmax<<<(N_NODES + 7) / 8, blk, 0, stream>>>(logits, out);
  }
}
